// PointerNet_16758962389318
// MI455X (gfx1250) — matrix operand order repaired and checked by reading, not compiled, not run on hardware
//
#include <hip/hip_runtime.h>
#include <math.h>

// PointerNet reference, CDNA5 (gfx1250, wave32) implementation.
// B=8, F=4000, S=512, E=H=128, TOPK=8, EPS=1e-4
//
// Pipeline:
//   K1: embedded[b,s,e] = tanh(inputs[b,:,s] . embedding[:,e])   (WMMA f16->f32, K=4000=125*32)
//       + global_prefetch of the next K chunk.
//   K2: probs[b,i,o]    = clip(sigmoid(embedded[b,i,:] . W[i,:,o] + bias[i,o]))  (WMMA)
//       W tiles staged to LDS via GLOBAL_LOAD_ASYNC_TO_LDS_B128 (ASYNCcnt) when
//       available, else coalesced float4 copies; also zero-fills prev_probs_2.
//   K3: sequential scan over S steps, one workgroup per batch, mask bitpacked in LDS.
//
// d_out layout (float32): [probs | zeros | prev_idxs | mask], each B*S*S.

#define BB   8
#define FF   4000
#define SS   512
#define EE   128
#define HH   128
#define TOPK 8
#define EPSV 1e-4f

typedef __attribute__((ext_vector_type(16))) _Float16 v16h;
typedef __attribute__((ext_vector_type(8)))  float    v8f;
typedef __attribute__((ext_vector_type(4)))  int      v4i;

typedef __attribute__((address_space(1))) v4i* gv4i_p;   // global int4*
typedef __attribute__((address_space(3))) v4i* lv4i_p;   // LDS int4*

#if defined(__has_builtin)
#if __has_builtin(__builtin_amdgcn_global_load_async_to_lds_b128)
#define USE_ASYNC_LDS 1
#else
#define USE_ASYNC_LDS 0
#endif
#else
#define USE_ASYNC_LDS 0
#endif

// ---------------------------------------------------------------------------
// Kernel 1: embedded = tanh(inputs^T @ embedding), per batch.
// Grid: (S/16, B) blocks; 256 threads = 8 waves; wave w owns N-tile w (E = 8*16).
// ---------------------------------------------------------------------------
__global__ __launch_bounds__(256) void k1_embed_tanh(
    const float* __restrict__ inputs,    // (B,F,S)
    const float* __restrict__ emb,       // (F,E)
    float* __restrict__ embedded)        // (B,S,E)
{
  const int stile = blockIdx.x;          // 0..31
  const int b     = blockIdx.y;          // 0..7
  const int tid   = threadIdx.x;
  const int wave  = tid >> 5;            // N-tile 0..7
  const int lane  = tid & 31;
  const int lm    = lane & 15;           // M (for A) / N (for B/C)
  const int half  = lane >> 4;
  const int s0    = stile * 16;
  const int n0    = wave * 16;

  const float* Ab = inputs + (size_t)b * FF * SS;
  v8f acc = {};

  for (int k0 = 0; k0 < FF; k0 += 32) {
    // prefetch next A chunk (lowers to global_prefetch_b8)
    if (k0 + 32 < FF)
      __builtin_prefetch(&Ab[(size_t)(k0 + 32) * SS + s0 + lm], 0, 0);

    v16h a, bf;
#pragma unroll
    for (int v = 0; v < 8; ++v) {
      // A (16x32 f16) per-lane K pairs: VGPR v<4 -> K=2v..; v>=4 -> K=16+2(v-4)..; +8*half
      const int kbA = ((v < 4) ? 2 * v : 16 + 2 * (v - 4)) + 8 * half;
      const int fA  = k0 + kbA;
      a[2 * v]     = (_Float16)Ab[(size_t)fA       * SS + s0 + lm];
      a[2 * v + 1] = (_Float16)Ab[(size_t)(fA + 1) * SS + s0 + lm];
      // B (32x16 f16): same K order as A: v<4 -> K=2v..; v>=4 -> K=16+2(v-4)..; +8*half
      const int kbB = ((v < 4) ? 2 * v : 16 + 2 * (v - 4)) + 8 * half;
      const int fB  = k0 + kbB;
      bf[2 * v]     = (_Float16)emb[(size_t)fB       * EE + n0 + lm];
      bf[2 * v + 1] = (_Float16)emb[(size_t)(fB + 1) * EE + n0 + lm];
    }
    acc = __builtin_amdgcn_wmma_f32_16x16x32_f16(false, a, false, bf,
                                                 (short)0, acc, false, false);
  }

  // C/D layout: lane half 0 -> M=v, half 1 -> M=v+8; N = lm
#pragma unroll
  for (int v = 0; v < 8; ++v) {
    const int M = v + 8 * half;
    embedded[((size_t)b * SS + (s0 + M)) * EE + n0 + lm] = tanhf(acc[v]);
  }
}

// ---------------------------------------------------------------------------
// Kernel 2: probs[b,i,o] = clip(sigmoid(embedded[b,i,:] @ W[i,:,o] + bias[i,o]))
// Grid: 512 blocks (one per i); 256 threads = 8 waves.
// W staged to LDS in 32(k) x 256(n) f32 tiles (32 KB), async when available.
// M dim = B = 8, padded to 16 (rows 8..15 zero). Each wave owns 2 tiles / n-half.
// ---------------------------------------------------------------------------
__global__ __launch_bounds__(256) void k2_logits_sigmoid(
    const float* __restrict__ embedded,  // (B,S,H)
    const float* __restrict__ W,         // (S,H,S)
    const float* __restrict__ bias,      // (S,S)
    float* __restrict__ probs,           // (B,S,S)
    float* __restrict__ zeros2)          // (B,S,S)
{
  __shared__ __align__(16) float Wtile[32 * 256];   // 32 KB

  const int i    = blockIdx.x;
  const int tid  = threadIdx.x;
  const int wave = tid >> 5;
  const int lane = tid & 31;
  const int lm   = lane & 15;
  const int half = lane >> 4;

  v8f acc[4] = {v8f{}, v8f{}, v8f{}, v8f{}};   // [nh*2 + j]

  for (int kk = 0; kk < HH / 32; ++kk) {
    const int k0 = kk * 32;

    // A fragment: rows 0..7 = embedded[b=M, i, k0..k0+31]; rows 8..15 = 0
    v16h a;
    const int M = lm;
#pragma unroll
    for (int v = 0; v < 8; ++v) {
      const int kbA = ((v < 4) ? 2 * v : 16 + 2 * (v - 4)) + 8 * half;
      if (M < BB) {
        const float* src = embedded + ((size_t)M * SS + i) * HH + k0 + kbA;
        a[2 * v]     = (_Float16)src[0];
        a[2 * v + 1] = (_Float16)src[1];
      } else {
        a[2 * v]     = (_Float16)0.0f;
        a[2 * v + 1] = (_Float16)0.0f;
      }
    }

    for (int nh = 0; nh < 2; ++nh) {
      // ---- stage W[i, k0..k0+31, nh*256..nh*256+255] into LDS (b128) ----
      __syncthreads();   // previous consumers done before overwrite
#pragma unroll
      for (int q = 0; q < 8; ++q) {
        const int idx4 = q * 256 + tid;          // 2048 float4 per tile
        const int row  = idx4 >> 6;              // 64 float4 per 256-float row
        const int c4   = idx4 & 63;
        const float* gp = W + ((size_t)i * HH + k0 + row) * SS + nh * 256 + c4 * 4;
        float*       lp = &Wtile[row * 256 + c4 * 4];
#if USE_ASYNC_LDS
        __builtin_amdgcn_global_load_async_to_lds_b128(
            (gv4i_p)const_cast<float*>(gp), (lv4i_p)lp, 0, 0);
#else
        *(float4*)lp = *(const float4*)gp;
#endif
      }
#if USE_ASYNC_LDS
#if __has_builtin(__builtin_amdgcn_s_wait_asynccnt)
      __builtin_amdgcn_s_wait_asynccnt(0);
#else
      asm volatile("s_wait_asynccnt 0" ::: "memory");
#endif
#endif
      __syncthreads();

      // ---- 2 tiles per wave from LDS ----
#pragma unroll
      for (int j = 0; j < 2; ++j) {
        const int colBase = (wave * 2 + j) * 16 + lm;   // local col in [0,256)
        v16h bf;
#pragma unroll
        for (int v = 0; v < 8; ++v) {
          const int kbB = ((v < 4) ? 2 * v : 16 + 2 * (v - 4)) + 8 * half;            // local row in [0,32)
          bf[2 * v]     = (_Float16)Wtile[kbB * 256 + colBase];
          bf[2 * v + 1] = (_Float16)Wtile[(kbB + 1) * 256 + colBase];
        }
        acc[nh * 2 + j] = __builtin_amdgcn_wmma_f32_16x16x32_f16(
            false, a, false, bf, (short)0, acc[nh * 2 + j], false, false);
      }
    }
  }

#pragma unroll
  for (int nh = 0; nh < 2; ++nh) {
#pragma unroll
    for (int j = 0; j < 2; ++j) {
      const int n0 = (nh * 16 + wave * 2 + j) * 16;
#pragma unroll
      for (int v = 0; v < 8; ++v) {
        const int M = v + 8 * half;       // = batch index
        if (M < BB) {
          const int n = n0 + lm;
          float x = acc[nh * 2 + j][v] + bias[(size_t)i * SS + n];
          float p = 1.0f / (1.0f + expf(-x));
          p = fminf(fmaxf(p, EPSV), 1.0f - EPSV);
          const size_t o = ((size_t)M * SS + i) * SS + n;
          probs[o]  = p;
          zeros2[o] = 0.0f;
        }
      }
    }
  }
}

// ---------------------------------------------------------------------------
// JAX threefry-2x32 (exact): 20 rounds, key schedule with 0x1BD11BDA parity.
// ---------------------------------------------------------------------------
__device__ __forceinline__ unsigned rotl32(unsigned x, int r) {
  return (x << r) | (x >> (32 - r));
}
__device__ __forceinline__ void threefry2x32(unsigned k0, unsigned k1,
                                             unsigned c0, unsigned c1,
                                             unsigned& o0, unsigned& o1) {
  const unsigned ks[3] = {k0, k1, k0 ^ k1 ^ 0x1BD11BDAu};
  unsigned x0 = c0 + ks[0], x1 = c1 + ks[1];
  const int R0[4] = {13, 15, 26, 6};
  const int R1[4] = {17, 29, 16, 24};
#pragma unroll
  for (int g = 0; g < 5; ++g) {
    const int* R = (g & 1) ? R1 : R0;
#pragma unroll
    for (int r = 0; r < 4; ++r) {
      x0 += x1;
      x1 = rotl32(x1, R[r]);
      x1 ^= x0;
    }
    x0 += ks[(g + 1) % 3];
    x1 += ks[(g + 2) % 3] + (unsigned)(g + 1);
  }
  o0 = x0; o1 = x1;
}

// ---------------------------------------------------------------------------
// Kernel 3: sequential scan (512 steps). One workgroup per batch, 512 threads.
// mask[b] bitpacked in LDS (512 rows x 16 u32 = 32 KB). Thread t owns row t
// and column t.
// ---------------------------------------------------------------------------
__global__ __launch_bounds__(512) void k3_select_mask(
    const float* __restrict__ probs,     // (B,S,S), already clipped
    float* __restrict__ prev_idxs,       // (B,S,S) as 0/1 float
    float* __restrict__ maskout)         // (B,S,S) as 0/1 float
{
  __shared__ unsigned      maskw[SS * 16];
  __shared__ float         pv[SS];
  __shared__ float         rv[SS];
  __shared__ int           ri[SS];
  __shared__ unsigned char sampb[SS];
  __shared__ unsigned char colb[SS];
  __shared__ unsigned char keepb[SS];
  __shared__ unsigned      sampw[16];
  __shared__ unsigned      roww[16];
  __shared__ unsigned      keyk[2];
  __shared__ int           winner;

  const int b = blockIdx.x;
  const int t = threadIdx.x;

#pragma unroll
  for (int w = 0; w < 16; ++w) maskw[t * 16 + w] = 0u;
  __syncthreads();

  for (int idx = 0; idx < SS; ++idx) {
    const float p = probs[((size_t)b * SS + idx) * SS + t];
    pv[t]    = p;
    keepb[t] = 0;
    if (t == 0) {
      // keys = split(key(42), S): threefry(k=(0,42), iota(1024)) -> (512,2)
      unsigned a0, a1, b0, b1;
      if (idx < 256) {
        threefry2x32(0u, 42u, 2u * idx,       2u * idx + 512u, a0, a1);
        threefry2x32(0u, 42u, 2u * idx + 1u,  2u * idx + 513u, b0, b1);
        keyk[0] = a0; keyk[1] = b0;          // x0-half outputs
      } else {
        threefry2x32(0u, 42u, 2u * idx - 512u, 2u * idx,       a0, a1);
        threefry2x32(0u, 42u, 2u * idx - 511u, 2u * idx + 1u,  b0, b1);
        keyk[0] = a1; keyk[1] = b1;          // x1-half outputs
      }
    }
    __syncthreads();

    // ---- top-8 (stable argmax x8; ties -> lower index, like lax.top_k) ----
    for (int r = 0; r < TOPK; ++r) {
      rv[t] = pv[t];
      ri[t] = t;
      __syncthreads();
      for (int str = 256; str > 0; str >>= 1) {
        if (t < str) {
          const float v2 = rv[t + str];
          const int   i2 = ri[t + str];
          if (v2 > rv[t] || (v2 == rv[t] && i2 < ri[t])) { rv[t] = v2; ri[t] = i2; }
        }
        __syncthreads();
      }
      if (t == 0) winner = ri[0];
      __syncthreads();
      if (t == winner) { keepb[t] = 1; pv[t] = -INFINITY; }
      __syncthreads();
    }

    // ---- bernoulli(keys[idx], p) over (B,S): linear j = b*512 + t ----
    const unsigned j  = (unsigned)(b * SS + t);
    const unsigned c0 = (j < 2048u) ? j : (j - 2048u);
    unsigned o0, o1;
    threefry2x32(keyk[0], keyk[1], c0, c0 + 2048u, o0, o1);
    const unsigned bits = (j < 2048u) ? o0 : o1;
    const float u = __uint_as_float((bits >> 9) | 0x3f800000u) - 1.0f;

    const int maskbit = (maskw[t * 16 + (idx >> 5)] >> (idx & 31)) & 1;
    const int s = (u < p) && keepb[t] && !maskbit && (t != idx);
    sampb[t] = (unsigned char)s;
    prev_idxs[((size_t)b * SS + idx) * SS + t] = s ? 1.0f : 0.0f;
    __syncthreads();

    if (t < 16) {
      unsigned wb = 0u;
#pragma unroll
      for (int q = 0; q < 32; ++q) wb |= ((unsigned)sampb[t * 32 + q]) << q;
      sampw[t] = wb;
    }
    __syncthreads();

    // anc_of_sel[a] = any_o(mask[a][o] & samp[o]); col[a] = mask[a][idx]|samp[a]|anc
    unsigned anc = 0u;
#pragma unroll
    for (int w = 0; w < 16; ++w) anc |= maskw[t * 16 + w] & sampw[w];
    const int colbit = maskbit | s | (anc != 0u);
    colb[t] = (unsigned char)colbit;

    // mask[:, idx] = col (colbit already includes old bit -> OR is exact)
    if (colbit) maskw[t * 16 + (idx >> 5)] |= 1u << (idx & 31);
    __syncthreads();

    // row = mask[idx, :] (after column update)
    if (t < 16) roww[t] = maskw[idx * 16 + t];
    __syncthreads();

    // mask |= col[:,None] & row[None,:]
    if (colb[t]) {
#pragma unroll
      for (int w = 0; w < 16; ++w) maskw[t * 16 + w] |= roww[w];
    }
    __syncthreads();
  }

  // ---- emit mask as 0/1 floats, coalesced over columns ----
  for (int k = 0; k < SS; ++k) {
    const unsigned bit = (maskw[k * 16 + (t >> 5)] >> (t & 31)) & 1u;
    maskout[((size_t)b * SS + k) * SS + t] = bit ? 1.0f : 0.0f;
  }
}

// ---------------------------------------------------------------------------
extern "C" void kernel_launch(void* const* d_in, const int* in_sizes, int n_in,
                              void* d_out, int out_size, void* d_ws, size_t ws_size,
                              hipStream_t stream) {
  const float* inputs    = (const float*)d_in[0];   // (B,F,S)
  const float* embedding = (const float*)d_in[1];   // (F,E)
  const float* W         = (const float*)d_in[2];   // (S,H,S)
  const float* bias      = (const float*)d_in[3];   // (S,S)
  // d_in[4] = offset_prob (unused by reference)

  float* out = (float*)d_out;
  const size_t N = (size_t)BB * SS * SS;
  float* probs    = out;
  float* probs2   = out + N;
  float* prevIdxs = out + 2 * N;
  float* maskOut  = out + 3 * N;

  float* embedded = (float*)d_ws;                   // B*S*E f32 = 2 MB scratch

  k1_embed_tanh<<<dim3(SS / 16, BB), 256, 0, stream>>>(inputs, embedding, embedded);
  k2_logits_sigmoid<<<dim3(SS), 256, 0, stream>>>(embedded, W, bias, probs, probs2);
  k3_select_mask<<<dim3(BB), 512, 0, stream>>>(probs, prevIdxs, maskOut);
}
